// RayCaster_77927886619273
// MI455X (gfx1250) — matrix operand order repaired and checked by reading, not compiled, not run on hardware
//
#include <hip/hip_runtime.h>
#include <hip/hip_bf16.h>
#include <math.h>

// ---------------------------------------------------------------------------
// NeRF two-pass ray caster for gfx1250 (MI455X).
// All GEMMs via v_wmma_f32_16x16x32_f16 (f16 in, f32 accumulate).
// M=64 points per wave: each B fragment feeds four WMMAs (minimal B traffic).
// Single in-place LDS activation buffer; A fragments fully hoisted to the
// 1024-VGPR wave32 register file before in-place stores.
// ---------------------------------------------------------------------------

typedef __attribute__((ext_vector_type(16))) _Float16 v16h;
typedef __attribute__((ext_vector_type(8)))  float    v8f;

#define TILE_M 64            // points per wave (four 16-row WMMA subtiles)
#define LDA 296              // sA row stride in halves (16B aligned, bank-safe)

union HFrag {
    v16h     v;
    _Float16 h[16];
    unsigned u[8];
    uint4    q[2];
};

// A fragment (16x32 f16) from an LDS row-major tile.
// ISA 7.12.2: lane L holds row M=L&15; halves 0..7 = k kb..kb+7 (contiguous),
// halves 8..15 = k kb+16..kb+23 -> two 16-byte LDS reads per lane.
__device__ __forceinline__ v16h load_a_lds(const _Float16* row, int ktile, int lane) {
    HFrag f;
    int kb = (ktile << 5) + ((lane & 16) ? 8 : 0);
#pragma unroll
    for (int i = 0; i < 8; ++i) {
        int k = kb + ((i & 4) << 2) + ((i & 3) << 1);
        f.u[i] = *(const unsigned*)(row + k);
    }
    return f.v;
}

// B fragment (32x16 f16) from transposed weights Wt[n][kpad] in global mem.
// Lane L: column n = (L&15)+16*ntile; halves 0..7 = k kb..kb+7, halves 8..15 =
// k kb+16..kb+23 with kb = ktile*32 + (L<16?0:8) -> two b128 loads per lane.
__device__ __forceinline__ v16h load_b_gl(const _Float16* __restrict__ Wt,
                                          int kpad, int ntile, int ktile, int lane) {
    int n  = (lane & 15) + (ntile << 4);
    int kb = (ktile << 5) + ((lane & 16) ? 8 : 0);
    const uint4* p = (const uint4*)(Wt + (size_t)n * kpad + kb);
    HFrag f;
    f.q[0] = p[0];
    f.q[1] = p[2];
    return f.v;
}

// One dense layer, IN-PLACE on the LDS tile:
//   base[64 x N] = relu(base[64 x K] @ W + b), K=32*KT, N=16*NT.
// All A fragments are hoisted to registers before any store, so reading and
// writing the same buffer is safe. Each B fragment serves MT(=4) WMMAs.
// Bias is folded into the accumulator init (C = broadcast bias per column).
// NOTE: deliberately no __restrict__ on `base` (loads must not sink past stores).
template <int KT, int NT, int MT>
__device__ __forceinline__ void mlp_layer_ip(_Float16* base,
                                             const _Float16* __restrict__ Wt, int kpad,
                                             const float* __restrict__ bias,
                                             int lane) {
    const int mloc = lane & 15;
    v16h a[MT][KT];
#pragma unroll
    for (int mi = 0; mi < MT; ++mi) {
        const _Float16* row = base + (size_t)(mi * 16 + mloc) * LDA;
#pragma unroll
        for (int kt = 0; kt < KT; ++kt) a[mi][kt] = load_a_lds(row, kt, lane);
    }
    const int mbase = (lane & 16) ? 8 : 0;
#pragma unroll
    for (int nt = 0; nt < NT; ++nt) {
        const int   n  = mloc + (nt << 4);
        const float bv = bias[n];
        v8f acc[MT];
#pragma unroll
        for (int mi = 0; mi < MT; ++mi)
#pragma unroll
            for (int r = 0; r < 8; ++r) acc[mi][r] = bv;
#pragma unroll
        for (int kt = 0; kt < KT; ++kt) {
            v16h b = load_b_gl(Wt, kpad, nt, kt, lane);
#pragma unroll
            for (int mi = 0; mi < MT; ++mi)
                acc[mi] = __builtin_amdgcn_wmma_f32_16x16x32_f16(
                    false, a[mi][kt], false, b, (short)0, acc[mi], false, false);
        }
#pragma unroll
        for (int mi = 0; mi < MT; ++mi)
#pragma unroll
            for (int r = 0; r < 8; ++r) {
                float x = fmaxf(acc[mi][r], 0.0f);
                base[(size_t)(mi * 16 + mbase + r) * LDA + n] = (_Float16)x;
            }
    }
}

// Narrow head (single 16-wide N tile): returns raw accumulators (bias pre-set).
template <int KT, int MT>
__device__ __forceinline__ void head_acc(const _Float16* base,
                                         const _Float16* __restrict__ Wt, int kpad,
                                         float binit, v8f* __restrict__ accOut, int lane) {
    const int mloc = lane & 15;
#pragma unroll
    for (int mi = 0; mi < MT; ++mi)
#pragma unroll
        for (int r = 0; r < 8; ++r) accOut[mi][r] = binit;
#pragma unroll
    for (int kt = 0; kt < KT; ++kt) {
        v16h b = load_b_gl(Wt, kpad, 0, kt, lane);
#pragma unroll
        for (int mi = 0; mi < MT; ++mi) {
            v16h a = load_a_lds(base + (size_t)(mi * 16 + mloc) * LDA, kt, lane);
            accOut[mi] = __builtin_amdgcn_wmma_f32_16x16x32_f16(
                false, a, false, b, (short)0, accOut[mi], false, false);
        }
    }
}

// ---------------------------------------------------------------------------
// Fused NeRF MLP: one wave per block owns 64 points. PE computed in-kernel
// (two rows per lane), activations staged in one in-place LDS tile.
// Layer flow inside sA: emb(0..63)+dir(256..287) -> h0(0..255) -> h1(0..255)
// -> h2(0..127); dir columns are never clobbered.
// ---------------------------------------------------------------------------
__global__ void __launch_bounds__(32)
mlp_kernel(const float* __restrict__ rb, const float* __restrict__ zbuf, int S,
           const _Float16* __restrict__ dirEmb,
           const _Float16* __restrict__ W0t, const _Float16* __restrict__ W1t,
           const _Float16* __restrict__ Wsigt, const _Float16* __restrict__ Wvt,
           const _Float16* __restrict__ Wrgbt,
           const float* __restrict__ b0, const float* __restrict__ b1,
           const float* __restrict__ bsig, const float* __restrict__ bvv,
           const float* __restrict__ brgb,
           float* __restrict__ raw) {
    __shared__ _Float16 sA[TILE_M][LDA];

    const int lane = threadIdx.x & 31;
    const int p0   = blockIdx.x * TILE_M;

    // ---- positional encoding: two full 64-wide rows per lane ----
#pragma unroll
    for (int q = 0; q < 2; ++q) {
        const int m   = lane + 32 * q;
        const int p   = p0 + m;
        const int ray = p / S;
        float        z   = zbuf[p];
        const float* rbr = rb + ray * 11;
        float pt[3];
#pragma unroll
        for (int c = 0; c < 3; ++c) pt[c] = rbr[c] + rbr[3 + c] * z;

        __attribute__((aligned(16))) _Float16 e[64];
        e[0] = (_Float16)pt[0];
        e[1] = (_Float16)pt[1];
        e[2] = (_Float16)pt[2];
        int idx = 3;
#pragma unroll
        for (int l = 0; l < 10; ++l) {
            float f = (float)(1 << l);
#pragma unroll
            for (int c = 0; c < 3; ++c) e[idx++] = (_Float16)__sinf(pt[c] * f);
#pragma unroll
            for (int c = 0; c < 3; ++c) e[idx++] = (_Float16)__cosf(pt[c] * f);
        }
        e[63] = (_Float16)0.0f;

        _Float16* rowA = &sA[m][0];
#pragma unroll
        for (int i = 0; i < 8; ++i) ((uint4*)rowA)[i] = ((const uint4*)e)[i];

        // direction embedding -> cols 256..287 (consumed by the Wv layer)
        const uint4* de4 = (const uint4*)(dirEmb + (size_t)ray * 32);
        uint4*       dst = (uint4*)(rowA + 256);
#pragma unroll
        for (int i = 0; i < 4; ++i) dst[i] = de4[i];
    }

    // ---- layer 0: h0 = relu(emb[64x64] @ W0 + b0), N=256 (in-place) ----
    mlp_layer_ip<2, 16, 4>(&sA[0][0], W0t, 64, b0, lane);
    // ---- layer 1: h1 = relu(h0 @ W1 + b1), K=256, N=256 (in-place) ----
    mlp_layer_ip<8, 16, 4>(&sA[0][0], W1t, 256, b1, lane);

    // ---- sigma head: h1 @ Wsig + bsig (only column n==0 is real) ----
    {
        v8f acc[4];
        head_acc<8, 4>(&sA[0][0], Wsigt, 256, bsig[0], acc, lane);
        if ((lane & 15) == 0) {
            int mb = (lane & 16) ? 8 : 0;
#pragma unroll
            for (int mi = 0; mi < 4; ++mi)
#pragma unroll
                for (int r = 0; r < 8; ++r)
                    raw[(size_t)(p0 + mi * 16 + mb + r) * 4 + 3] = acc[mi][r];
        }
    }

    // ---- view layer: h2 = relu([h1|dir][64x288] @ Wv + bv), N=128 (in-place) ----
    mlp_layer_ip<9, 8, 4>(&sA[0][0], Wvt, 288, bvv, lane);

    // ---- rgb head: h2 @ Wrgb + brgb (columns n<3 real) ----
    {
        int n = lane & 15;
        v8f acc[4];
        head_acc<4, 4>(&sA[0][0], Wrgbt, 128, brgb[n < 3 ? n : 2], acc, lane);
        if (n < 3) {
            int mb = (lane & 16) ? 8 : 0;
#pragma unroll
            for (int mi = 0; mi < 4; ++mi)
#pragma unroll
                for (int r = 0; r < 8; ++r)
                    raw[(size_t)(p0 + mi * 16 + mb + r) * 4 + n] = acc[mi][r];
        }
    }
}

// ---------------------------------------------------------------------------
// Weight conversion: f32 W[Kreal][Nreal] -> f16 Wt[Npad][Kpad] (transposed,
// zero-padded) so B fragments are two 16-byte reads per lane.
// ---------------------------------------------------------------------------
__global__ void convert_wt(const float* __restrict__ W, _Float16* __restrict__ Wt,
                           int Kreal, int Nreal, int Kpad, int Npad) {
    int i     = blockIdx.x * blockDim.x + threadIdx.x;
    int total = Npad * Kpad;
    if (i >= total) return;
    int   n = i / Kpad;
    int   k = i - n * Kpad;
    float v = (k < Kreal && n < Nreal) ? W[(size_t)k * Nreal + n] : 0.0f;
    Wt[i] = (_Float16)v;
}

__global__ void zvals_kernel(const float* __restrict__ rb, float* __restrict__ z, int S) {
    int   i    = blockIdx.x * blockDim.x + threadIdx.x;
    int   ray  = i / S;
    int   s    = i - ray * S;
    float near = rb[ray * 11 + 6];
    float far  = rb[ray * 11 + 7];
    z[i] = near + (far - near) * ((float)s / (float)(S - 1));
}

__global__ void diremb_kernel(const float* __restrict__ rb, _Float16* __restrict__ de) {
    int          ray = blockIdx.x * blockDim.x + threadIdx.x;
    const float* v   = rb + ray * 11 + 8;
    float        d0 = v[0], d1 = v[1], d2 = v[2];
    _Float16*    o  = de + (size_t)ray * 32;
    o[0] = (_Float16)d0; o[1] = (_Float16)d1; o[2] = (_Float16)d2;
    int idx = 3;
#pragma unroll
    for (int l = 0; l < 4; ++l) {
        float f = (float)(1 << l);
        o[idx++] = (_Float16)__sinf(d0 * f);
        o[idx++] = (_Float16)__sinf(d1 * f);
        o[idx++] = (_Float16)__sinf(d2 * f);
        o[idx++] = (_Float16)__cosf(d0 * f);
        o[idx++] = (_Float16)__cosf(d1 * f);
        o[idx++] = (_Float16)__cosf(d2 * f);
    }
    for (; idx < 32; ++idx) o[idx] = (_Float16)0.0f;
}

// Coarse compositing + importance resampling (sample_pdf) + sorted merge.
__global__ void composite_coarse_kernel(const float* __restrict__ rb,
                                        const float* __restrict__ zc,
                                        const float* __restrict__ raw,
                                        float* __restrict__ rgb0,
                                        float* __restrict__ zall) {
    const int    ray = blockIdx.x * blockDim.x + threadIdx.x;
    const int    S   = 64;
    const float* z   = zc + (size_t)ray * S;
    const float* rw  = raw + (size_t)ray * S * 4;

    float dn;
    {
        const float* d = rb + ray * 11 + 3;
        dn = sqrtf(d[0] * d[0] + d[1] * d[1] + d[2] * d[2]);
    }
    float wts[64];
    float acc0 = 0.f, acc1 = 0.f, acc2 = 0.f;
    float T = 1.0f;
    for (int i = 0; i < S; ++i) {
        float dist  = ((i < S - 1) ? (z[i + 1] - z[i]) : 1e10f) * dn;
        float sg    = fmaxf(rw[i * 4 + 3], 0.0f);
        float alpha = 1.0f - __expf(-sg * dist);
        float wt    = alpha * T;
        wts[i] = wt;
        acc0 += wt / (1.0f + __expf(-rw[i * 4 + 0]));
        acc1 += wt / (1.0f + __expf(-rw[i * 4 + 1]));
        acc2 += wt / (1.0f + __expf(-rw[i * 4 + 2]));
        T *= (1.0f - alpha + 1e-10f);
    }
    rgb0[ray * 3 + 0] = acc0;
    rgb0[ray * 3 + 1] = acc1;
    rgb0[ray * 3 + 2] = acc2;

    // ---- sample_pdf over weights[1:-1] with bins = z midpoints ----
    float bins[63];
    for (int i = 0; i < 63; ++i) bins[i] = 0.5f * (z[i] + z[i + 1]);
    float pdf[62];
    float sum = 0.0f;
    for (int i = 0; i < 62; ++i) {
        pdf[i] = wts[i + 1] + 1e-5f;
        sum += pdf[i];
    }
    float inv = 1.0f / sum;
    float cdf[63];
    cdf[0] = 0.0f;
    for (int i = 0; i < 62; ++i) cdf[i + 1] = cdf[i] + pdf[i] * inv;

    float zs[64];
    for (int j = 0; j < 64; ++j) {
        float u   = (float)j / 63.0f;
        int   ind = 63;                       // searchsorted 'right'
        for (int i = 0; i < 63; ++i) {
            if (cdf[i] > u) { ind = i; break; }
        }
        int   below = min(max(ind - 1, 0), 62);
        int   above = min(ind, 62);
        float cb = cdf[below], ca = cdf[above];
        float denom = (ca - cb < 1e-5f) ? 1.0f : (ca - cb);
        float t     = (u - cb) / denom;
        zs[j] = bins[below] + t * (bins[above] - bins[below]);
    }

    // merge two sorted lists -> z_all[128]
    float* za = zall + (size_t)ray * 128;
    int a = 0, b = 0;
    for (int i = 0; i < 128; ++i) {
        float va = (a < 64) ? z[a]  : 3.4e38f;
        float vb = (b < 64) ? zs[b] : 3.4e38f;
        if (va <= vb) { za[i] = va; ++a; }
        else          { za[i] = vb; ++b; }
    }
}

__global__ void composite_fine_kernel(const float* __restrict__ rb,
                                      const float* __restrict__ zall,
                                      const float* __restrict__ raw,
                                      const float* __restrict__ rgb0,
                                      float* __restrict__ out) {
    const int    ray = blockIdx.x * blockDim.x + threadIdx.x;
    const int    S   = 128;
    const float* z   = zall + (size_t)ray * S;
    const float* rw  = raw + (size_t)ray * S * 4;

    float dn;
    {
        const float* d = rb + ray * 11 + 3;
        dn = sqrtf(d[0] * d[0] + d[1] * d[1] + d[2] * d[2]);
    }
    float acc0 = 0.f, acc1 = 0.f, acc2 = 0.f;
    float T = 1.0f;
    for (int i = 0; i < S; ++i) {
        float dist  = ((i < S - 1) ? (z[i + 1] - z[i]) : 1e10f) * dn;
        float sg    = fmaxf(rw[i * 4 + 3], 0.0f);
        float alpha = 1.0f - __expf(-sg * dist);
        float wt    = alpha * T;
        acc0 += wt / (1.0f + __expf(-rw[i * 4 + 0]));
        acc1 += wt / (1.0f + __expf(-rw[i * 4 + 1]));
        acc2 += wt / (1.0f + __expf(-rw[i * 4 + 2]));
        T *= (1.0f - alpha + 1e-10f);
    }
    out[ray * 6 + 0] = acc0;
    out[ray * 6 + 1] = acc1;
    out[ray * 6 + 2] = acc2;
    out[ray * 6 + 3] = rgb0[ray * 3 + 0];
    out[ray * 6 + 4] = rgb0[ray * 3 + 1];
    out[ray * 6 + 5] = rgb0[ray * 3 + 2];
}

// ---------------------------------------------------------------------------
// Host-side orchestration.
// Input order (JAX sorted-key dict flattening):
//   0: ray_batch
//   params_c: 1:W0 2:W1 3:Wrgb 4:Wsig 5:Wv 6:b0 7:b1 8:brgb 9:bsig 10:bv
//   params_f: 11..20 (same order)
//   21: N_samples(=64)  22: N_importance(=64)
// ---------------------------------------------------------------------------
extern "C" void kernel_launch(void* const* d_in, const int* in_sizes, int n_in,
                              void* d_out, int out_size, void* d_ws, size_t ws_size,
                              hipStream_t stream) {
    const float* rb     = (const float*)d_in[0];
    const int    n_rays = in_sizes[0] / 11;
    const int    SC = 64, SF = 128;

    // ---- workspace carve-up ----
    size_t off = 0;
    auto alloc = [&](size_t bytes) -> void* {
        off = (off + 255) & ~(size_t)255;
        void* p = (char*)d_ws + off;
        off += bytes;
        return p;
    };
    auto allocH = [&](size_t halves) -> _Float16* { return (_Float16*)alloc(halves * 2); };
    auto allocF = [&](size_t n) -> float* { return (float*)alloc(n * 4); };

    _Float16* Wt[2][5];   // per net: W0t, W1t, Wsigt, Wvt, Wrgbt
    for (int net = 0; net < 2; ++net) {
        Wt[net][0] = allocH(256 * 64);    // W0t  [256][64]
        Wt[net][1] = allocH(256 * 256);   // W1t  [256][256]
        Wt[net][2] = allocH(16 * 256);    // Wsigt[16][256]
        Wt[net][3] = allocH(128 * 288);   // Wvt  [128][288]
        Wt[net][4] = allocH(16 * 128);    // Wrgbt[16][128]
    }
    _Float16* dirEmb = allocH((size_t)n_rays * 32);
    float*    z_c    = allocF((size_t)n_rays * SC);
    float*    z_all  = allocF((size_t)n_rays * SF);
    float*    raw_c  = allocF((size_t)n_rays * SC * 4);
    float*    raw_f  = allocF((size_t)n_rays * SF * 4);
    float*    rgb0   = allocF((size_t)n_rays * 3);

    // ---- weight conversion (K, Nreal, Kpad, Npad), input index per net ----
    const int convK[5]    = {63, 256, 256, 283, 128};
    const int convN[5]    = {256, 256, 1, 128, 3};
    const int convKpad[5] = {64, 256, 256, 288, 128};
    const int convNpad[5] = {256, 256, 16, 128, 16};
    const int convIdx[5]  = {1, 2, 4, 5, 3}; // W0, W1, Wsig, Wv, Wrgb in sorted order
    for (int net = 0; net < 2; ++net) {
        int base = net * 10;
        for (int j = 0; j < 5; ++j) {
            int total = convNpad[j] * convKpad[j];
            hipLaunchKernelGGL(convert_wt, dim3((total + 255) / 256), dim3(256), 0, stream,
                               (const float*)d_in[base + convIdx[j]], Wt[net][j],
                               convK[j], convN[j], convKpad[j], convNpad[j]);
        }
    }

    // ---- coarse z + direction embedding ----
    hipLaunchKernelGGL(zvals_kernel, dim3((n_rays * SC) / 256), dim3(256), 0, stream,
                       rb, z_c, SC);
    hipLaunchKernelGGL(diremb_kernel, dim3((n_rays + 255) / 256), dim3(256), 0, stream,
                       rb, dirEmb);

    // ---- coarse MLP ----
    {
        int P      = n_rays * SC;
        int blocks = P / TILE_M;
        hipLaunchKernelGGL(mlp_kernel, dim3(blocks), dim3(32), 0, stream,
                           rb, z_c, SC, dirEmb,
                           Wt[0][0], Wt[0][1], Wt[0][2], Wt[0][3], Wt[0][4],
                           (const float*)d_in[6], (const float*)d_in[7],
                           (const float*)d_in[9], (const float*)d_in[10],
                           (const float*)d_in[8], raw_c);
    }

    // ---- composite coarse + importance sampling + merge ----
    hipLaunchKernelGGL(composite_coarse_kernel, dim3((n_rays + 255) / 256), dim3(256),
                       0, stream, rb, z_c, raw_c, rgb0, z_all);

    // ---- fine MLP ----
    {
        int P      = n_rays * SF;
        int blocks = P / TILE_M;
        hipLaunchKernelGGL(mlp_kernel, dim3(blocks), dim3(32), 0, stream,
                           rb, z_all, SF, dirEmb,
                           Wt[1][0], Wt[1][1], Wt[1][2], Wt[1][3], Wt[1][4],
                           (const float*)d_in[16], (const float*)d_in[17],
                           (const float*)d_in[19], (const float*)d_in[20],
                           (const float*)d_in[18], raw_f);
    }

    // ---- composite fine -> output [rgb_f | rgb0] ----
    hipLaunchKernelGGL(composite_fine_kernel, dim3((n_rays + 255) / 256), dim3(256),
                       0, stream, rb, z_all, raw_f, rgb0, (float*)d_out);
}
